// teanet_57861799412347
// MI455X (gfx1250) — matrix operand order repaired and checked by reading, not compiled, not run on hardware
//
#include <hip/hip_runtime.h>

// ---------------------------------------------------------------------------
// TeaNet forward for MI455X (gfx1250, wave32, WMMA).
// Heavy GEMMs (32768x1024x1024) run on v_wmma_f32_16x16x32_bf16 with fp32
// accumulation; activations/weights are converted once to bf16.
// GEMM producer loop uses gfx1250 async global->LDS copies (ASYNCcnt) with
// double-buffered LDS when the toolchain exposes the builtins.
// ---------------------------------------------------------------------------

typedef __bf16 bf16_t;
typedef __attribute__((ext_vector_type(16))) __bf16 bf16x16;
typedef __attribute__((ext_vector_type(8)))  float  v8f;
typedef __attribute__((ext_vector_type(4)))  unsigned int uint4v;

union Frag { bf16x16 v; uint4v q[2]; };

#define LAG   8192
#define KTW   4
#define DIM   1024
#define HEADS 16
#define DH    64
#define HID   5
#define MROWS (LAG * KTW)          // 32768

#if defined(__has_builtin)
#if __has_builtin(__builtin_amdgcn_global_load_async_to_lds_b128)
#define HAVE_ASYNC_LDS 1
#endif
#endif

#ifdef HAVE_ASYNC_LDS
// builtin signature (from hipcc diagnostic): first param is AS(1) pointer to
// int __attribute__((vector_size(16))), non-const; LDS side is AS(3).
typedef int v4i_n __attribute__((vector_size(16)));
typedef __attribute__((address_space(1))) v4i_n* as1_v4i;
typedef __attribute__((address_space(3))) v4i_n* as3_v4i;
#endif

__device__ __forceinline__ void cp_tile16(bf16_t* ldst, const bf16_t* gsrc) {
#ifdef HAVE_ASYNC_LDS
  // memory -> LDS without touching VGPRs; tracked by ASYNCcnt
  bf16_t* g = const_cast<bf16_t*>(gsrc);
  __builtin_amdgcn_global_load_async_to_lds_b128(
      (as1_v4i)g, (as3_v4i)ldst, 0, 0);
#else
  *(uint4v*)ldst = *(const uint4v*)gsrc;
#endif
}

__device__ __forceinline__ void async_tiles_wait() {
#ifdef HAVE_ASYNC_LDS
#if __has_builtin(__builtin_amdgcn_s_wait_asynccnt)
  __builtin_amdgcn_s_wait_asynccnt(0);
#else
  asm volatile("s_wait_asynccnt 0" ::: "memory");
#endif
#endif
}

__device__ __forceinline__ float wred_sum(float v) {
  for (int o = 16; o > 0; o >>= 1) v += __shfl_xor(v, o, 32);
  return v;
}
__device__ __forceinline__ float sigm(float x) { return 1.f / (1.f + expf(-x)); }

// --------------------------- prepass kernels --------------------------------

// W [K=1024][N=1024] f32 -> BT [N][K] bf16 (transposed so K is contiguous)
__global__ void k_wt_bt(const float* __restrict__ W, bf16_t* __restrict__ BT) {
  int t = blockIdx.x * blockDim.x + threadIdx.x;      // 0..2^20-1
  int k = t >> 10, n = t & 1023;
  BT[(size_t)n * DIM + k] = (bf16_t)W[t];
}

// X = text + pos_embed  (f32 -> bf16)
__global__ void k_add_pos(const float* __restrict__ a, const float* __restrict__ b,
                          bf16_t* __restrict__ X) {
  size_t t = (size_t)blockIdx.x * blockDim.x + threadIdx.x;
  X[t] = (bf16_t)(a[t] + b[t]);
}

// ------------------------------- WMMA GEMM ----------------------------------
// C[M][1024] = A[M][1024] @ BT^T  (BT stored [N][K], K contiguous)
// epi: 0 = none, 1 = +bias, 2 = +bias then ReLU
#define TM  128
#define TN  128
#define TK  32
#define LDT 40    // padded LDS row stride in bf16 elems (80B, 16B aligned, no bank conflicts)

__global__ __launch_bounds__(256)
void k_gemm(const bf16_t* __restrict__ A, const bf16_t* __restrict__ BT,
            const float* __restrict__ bias, bf16_t* __restrict__ C, int epi) {
  __shared__ bf16_t lA[2][TM * LDT];   // double-buffered: 2 x 10 KB
  __shared__ bf16_t lB[2][TN * LDT];

  const int tid   = threadIdx.x;
  const int lane  = tid & 31;
  const int wave  = tid >> 5;
  const int waveM = wave & 3;          // 4 waves along M
  const int waveN = wave >> 2;         // 2 waves along N
  const int lm    = lane & 15;
  const int sel   = lane >> 4;         // 0: lanes 0-15, 1: lanes 16-31
  const int blockM = blockIdx.y * TM;
  const int blockN = blockIdx.x * TN;

  // two 16B chunks per thread per matrix per K-step (128 rows x 4 segs = 512 chunks)
  const int r0 = tid >> 2,          s0 = tid & 3;
  const int r1 = (tid + 256) >> 2,  s1 = (tid + 256) & 3;

  const bf16_t* gA0 = A  + (size_t)(blockM + r0) * DIM + s0 * 8;
  const bf16_t* gA1 = A  + (size_t)(blockM + r1) * DIM + s1 * 8;
  const bf16_t* gB0 = BT + (size_t)(blockN + r0) * DIM + s0 * 8;
  const bf16_t* gB1 = BT + (size_t)(blockN + r1) * DIM + s1 * 8;
  const int dA0 = r0 * LDT + s0 * 8, dA1 = r1 * LDT + s1 * 8;
  const int dB0 = r0 * LDT + s0 * 8, dB1 = r1 * LDT + s1 * 8;

  v8f acc[2][4] = {};

  // prologue: stage tile 0 into buffer 0
  cp_tile16(&lA[0][dA0], gA0);
  cp_tile16(&lA[0][dA1], gA1);
  cp_tile16(&lB[0][dB0], gB0);
  cp_tile16(&lB[0][dB1], gB1);

  for (int k0 = 0; k0 < DIM; k0 += TK) {
    const int cur = (k0 >> 5) & 1;
    async_tiles_wait();     // my async copies for buffer `cur` are complete
    __syncthreads();        // everyone's copies complete; prior reads of 1-cur done

    // stage next K tile into the other buffer while we compute on this one
    if (k0 + TK < DIM) {
      const int nxt = 1 - cur;
      cp_tile16(&lA[nxt][dA0], gA0 + k0 + TK);
      cp_tile16(&lA[nxt][dA1], gA1 + k0 + TK);
      cp_tile16(&lB[nxt][dB0], gB0 + k0 + TK);
      cp_tile16(&lB[nxt][dB1], gB1 + k0 + TK);
      if (k0 + 2 * TK < DIM) {   // gfx1250 global_prefetch_b8 two tiles ahead
        __builtin_prefetch(gA0 + k0 + 2 * TK, 0, 1);
        __builtin_prefetch(gB0 + k0 + 2 * TK, 0, 1);
      }
    }

    // Fragment gathers per CDNA5 ISA bf16 layouts:
    //  A 16x32: lane<16 -> row lm, K{0..7} then K{16..23}; lane>=16 -> K{8..15},K{24..31}
    //  B 32x16: lane<16 -> col lm, K{0..7} then K{16..23}; lane>=16 -> K{8..15},K{24..31}
    Frag a[2], b[4];
#pragma unroll
    for (int mt = 0; mt < 2; ++mt) {
      const int row = waveM * 32 + mt * 16 + lm;
      a[mt].q[0] = *(const uint4v*)(&lA[cur][row * LDT + sel * 8]);
      a[mt].q[1] = *(const uint4v*)(&lA[cur][row * LDT + 16 + sel * 8]);
    }
#pragma unroll
    for (int nt = 0; nt < 4; ++nt) {
      const int col = waveN * 64 + nt * 16 + lm;
      b[nt].q[0] = *(const uint4v*)(&lB[cur][col * LDT + sel * 8]);
      b[nt].q[1] = *(const uint4v*)(&lB[cur][col * LDT + 16 + sel * 8]);
    }
#pragma unroll
    for (int mt = 0; mt < 2; ++mt)
#pragma unroll
      for (int nt = 0; nt < 4; ++nt)
        acc[mt][nt] = __builtin_amdgcn_wmma_f32_16x16x32_bf16(
            false, a[mt].v, false, b[nt].v, (short)0, acc[mt][nt], false, false);
  }

  // epilogue: C/D layout -> lane gives N (lm), VGPR r gives M=r (+8 for lanes>=16)
#pragma unroll
  for (int mt = 0; mt < 2; ++mt)
#pragma unroll
    for (int nt = 0; nt < 4; ++nt) {
      const int n = blockN + waveN * 64 + nt * 16 + lm;
      const float bv = (epi != 0) ? bias[n] : 0.f;
#pragma unroll
      for (int r = 0; r < 8; ++r) {
        const int m = blockM + waveM * 32 + mt * 16 + r + sel * 8;
        float val = acc[mt][nt][r] + bv;
        if (epi == 2) val = fmaxf(val, 0.f);
        C[(size_t)m * DIM + n] = (bf16_t)val;
      }
    }
}

// ----------------------- attention / pointwise kernels ----------------------

// multi-head attention over k=4, dh=64; one wave per (lag, head)
__global__ void k_attn_mh(const bf16_t* __restrict__ Qp, const bf16_t* __restrict__ Kp,
                          const bf16_t* __restrict__ Vp, bf16_t* __restrict__ Op) {
  const int gw   = blockIdx.x * (blockDim.x >> 5) + (threadIdx.x >> 5);
  const int lane = threadIdx.x & 31;
  const int l = gw >> 4, h = gw & 15;
  const size_t rb = (size_t)(l * KTW) * DIM + h * DH;

  float q[4][2], kk[4][2], vv[4][2];
#pragma unroll
  for (int i = 0; i < 4; ++i) {
    const size_t o = rb + (size_t)i * DIM;
    q[i][0]  = (float)Qp[o + lane]; q[i][1]  = (float)Qp[o + lane + 32];
    kk[i][0] = (float)Kp[o + lane]; kk[i][1] = (float)Kp[o + lane + 32];
    vv[i][0] = (float)Vp[o + lane]; vv[i][1] = (float)Vp[o + lane + 32];
  }
  float a[4][4];
#pragma unroll
  for (int i = 0; i < 4; ++i)
#pragma unroll
    for (int j = 0; j < 4; ++j) {
      float s = q[i][0] * kk[j][0] + q[i][1] * kk[j][1];
      a[i][j] = wred_sum(s) * 0.125f;   // 1/sqrt(64)
    }
#pragma unroll
  for (int i = 0; i < 4; ++i) {
    float mx = fmaxf(fmaxf(a[i][0], a[i][1]), fmaxf(a[i][2], a[i][3]));
    float s = 0.f;
#pragma unroll
    for (int j = 0; j < 4; ++j) { a[i][j] = expf(a[i][j] - mx); s += a[i][j]; }
    const float inv = 1.f / s;
#pragma unroll
    for (int j = 0; j < 4; ++j) a[i][j] *= inv;
  }
#pragma unroll
  for (int i = 0; i < 4; ++i) {
    float o0 = 0.f, o1 = 0.f;
#pragma unroll
    for (int j = 0; j < 4; ++j) { o0 += a[i][j] * vv[j][0]; o1 += a[i][j] * vv[j][1]; }
    const size_t o = rb + (size_t)i * DIM;
    Op[o + lane] = (bf16_t)o0; Op[o + lane + 32] = (bf16_t)o1;
  }
}

// single-head attention over k=4, dh=1024; one 256-thread block per lag
__global__ void k_attn_sh(const bf16_t* __restrict__ Qp, const bf16_t* __restrict__ Kp,
                          const bf16_t* __restrict__ Vp, bf16_t* __restrict__ Op) {
  __shared__ float sred[16];
  const int l = blockIdx.x, t = threadIdx.x;
  if (t < 16) sred[t] = 0.f;
  __syncthreads();
  const size_t base = (size_t)l * KTW * DIM;

  float part[16];
#pragma unroll
  for (int c = 0; c < 16; ++c) part[c] = 0.f;
  for (int r = 0; r < 4; ++r) {
    const int d = t + 256 * r;
    float qv[4], kv[4];
#pragma unroll
    for (int i = 0; i < 4; ++i) {
      qv[i] = (float)Qp[base + (size_t)i * DIM + d];
      kv[i] = (float)Kp[base + (size_t)i * DIM + d];
    }
#pragma unroll
    for (int i = 0; i < 4; ++i)
#pragma unroll
      for (int j = 0; j < 4; ++j) part[i * 4 + j] += qv[i] * kv[j];
  }
#pragma unroll
  for (int c = 0; c < 16; ++c) {
    float s = wred_sum(part[c]);
    if ((t & 31) == 0) atomicAdd(&sred[c], s);
  }
  __syncthreads();

  float a[16];
#pragma unroll
  for (int i = 0; i < 4; ++i) {
    float s0 = sred[i * 4 + 0] * 0.03125f, s1 = sred[i * 4 + 1] * 0.03125f;
    float s2 = sred[i * 4 + 2] * 0.03125f, s3 = sred[i * 4 + 3] * 0.03125f;
    float mx = fmaxf(fmaxf(s0, s1), fmaxf(s2, s3));
    float e0 = expf(s0 - mx), e1 = expf(s1 - mx), e2 = expf(s2 - mx), e3 = expf(s3 - mx);
    float inv = 1.f / (e0 + e1 + e2 + e3);
    a[i * 4 + 0] = e0 * inv; a[i * 4 + 1] = e1 * inv;
    a[i * 4 + 2] = e2 * inv; a[i * 4 + 3] = e3 * inv;
  }
  for (int r = 0; r < 4; ++r) {
    const int d = t + 256 * r;
    float vv[4];
#pragma unroll
    for (int j = 0; j < 4; ++j) vv[j] = (float)Vp[base + (size_t)j * DIM + d];
#pragma unroll
    for (int i = 0; i < 4; ++i) {
      float o = a[i*4+0]*vv[0] + a[i*4+1]*vv[1] + a[i*4+2]*vv[2] + a[i*4+3]*vv[3];
      Op[base + (size_t)i * DIM + d] = (bf16_t)o;
    }
  }
}

// Y = LayerNorm(inter + x) * g + b ; one wave per row of 1024
__global__ void k_ln(const bf16_t* __restrict__ inter, const bf16_t* __restrict__ x,
                     const float* __restrict__ g, const float* __restrict__ b,
                     bf16_t* __restrict__ Y) {
  const int row  = blockIdx.x * 8 + (threadIdx.x >> 5);
  const int lane = threadIdx.x & 31;
  const size_t base = (size_t)row * DIM;
  float v[32], s = 0.f, s2 = 0.f;
#pragma unroll
  for (int j = 0; j < 32; ++j) {
    const int d = lane + 32 * j;
    float t = (float)inter[base + d] + (float)x[base + d];
    v[j] = t; s += t; s2 += t * t;
  }
  for (int o = 16; o > 0; o >>= 1) { s += __shfl_xor(s, o, 32); s2 += __shfl_xor(s2, o, 32); }
  const float mu = s * (1.f / 1024.f);
  const float var = s2 * (1.f / 1024.f) - mu * mu;
  const float rs = rsqrtf(var + 1e-5f);
#pragma unroll
  for (int j = 0; j < 32; ++j) {
    const int d = lane + 32 * j;
    Y[base + d] = (bf16_t)((v[j] - mu) * rs * g[d] + b[d]);
  }
}

// att = softmax(w_u^T @ tanh(w_m)) ; single 1024-thread block
__global__ void k_attvec(const float* __restrict__ wu, const float* __restrict__ wm,
                         float* __restrict__ att) {
  __shared__ float sh[32];
  __shared__ float stat[2];
  const int j = threadIdx.x;
  float acc = 0.f;
  for (int d = 0; d < DIM; ++d) acc += wu[(size_t)d * DIM + j] * tanhf(wm[d]);
  float m = acc;
  for (int o = 16; o > 0; o >>= 1) m = fmaxf(m, __shfl_xor(m, o, 32));
  if ((j & 31) == 0) sh[j >> 5] = m;
  __syncthreads();
  if (j < 32) {
    float v2 = sh[j];
    for (int o = 16; o > 0; o >>= 1) v2 = fmaxf(v2, __shfl_xor(v2, o, 32));
    if (j == 0) stat[0] = v2;
  }
  __syncthreads();
  const float e = expf(acc - stat[0]);
  float s = e;
  for (int o = 16; o > 0; o >>= 1) s += __shfl_xor(s, o, 32);
  __syncthreads();
  if ((j & 31) == 0) sh[j >> 5] = s;
  __syncthreads();
  if (j < 32) {
    float v2 = sh[j];
    for (int o = 16; o > 0; o >>= 1) v2 += __shfl_xor(v2, o, 32);
    if (j == 0) stat[1] = v2;
  }
  __syncthreads();
  att[j] = e / stat[1];
}

// out_k[l][i] = M[l*4+i][:] . att  -> lstm_in[l][0..3] ; one wave per row
__global__ void k_outk(const bf16_t* __restrict__ Mm, const float* __restrict__ att,
                       float* __restrict__ lstm_in) {
  const int row  = blockIdx.x * 8 + (threadIdx.x >> 5);
  const int lane = threadIdx.x & 31;
  const bf16_t* p = Mm + (size_t)row * DIM;
  float acc = 0.f;
#pragma unroll
  for (int j = 0; j < 32; ++j) { const int d = lane + 32 * j; acc += (float)p[d] * att[d]; }
  acc = wred_sum(acc);
  if (lane == 0) { const int l = row >> 2, i = row & 3; lstm_in[l * 9 + i] = acc; }
}

__global__ void k_prices(const float* __restrict__ prices, float* __restrict__ lstm_in) {
  const int t = blockIdx.x * blockDim.x + threadIdx.x;
  if (t >= LAG * 5) return;
  const int l = t / 5, p = t % 5;
  lstm_in[l * 9 + 4 + p] = prices[t];
}

// sequential LSTM, 1 wave; lane g<20 owns gate g, lanes 0..4 own h/c
__global__ void k_lstm(const float* __restrict__ xin, const float* __restrict__ Wih,
                       const float* __restrict__ Whh, const float* __restrict__ bih,
                       const float* __restrict__ bhh, float* __restrict__ hs) {
  const int lane = threadIdx.x;
  float wih[9], whh[5], bsum = 0.f;
  if (lane < 20) {
#pragma unroll
    for (int x = 0; x < 9; ++x) wih[x] = Wih[lane * 9 + x];
#pragma unroll
    for (int j = 0; j < 5; ++j) whh[j] = Whh[lane * 5 + j];
    bsum = bih[lane] + bhh[lane];
  }
  float h[5] = {0.f, 0.f, 0.f, 0.f, 0.f};
  float c = 0.f;
  const int j5 = (lane < 5) ? lane : 0;
  for (int t = 0; t < LAG; ++t) {
    const float* xt = xin + t * 9;
    float g = bsum;
    if (lane < 20) {
#pragma unroll
      for (int x = 0; x < 9; ++x) g += wih[x] * xt[x];
#pragma unroll
      for (int j = 0; j < 5; ++j) g += whh[j] * h[j];
    }
    const float iv = __shfl(g, j5,      32);
    const float fv = __shfl(g, j5 + 5,  32);
    const float gg = __shfl(g, j5 + 10, 32);
    const float ov = __shfl(g, j5 + 15, 32);
    float hn = 0.f;
    if (lane < 5) {
      c  = sigm(fv) * c + sigm(iv) * tanhf(gg);
      hn = sigm(ov) * tanhf(c);
      hs[t * 5 + lane] = hn;
    }
    h[0] = __shfl(hn, 0, 32); h[1] = __shfl(hn, 1, 32); h[2] = __shfl(hn, 2, 32);
    h[3] = __shfl(hn, 3, 32); h[4] = __shfl(hn, 4, 32);
  }
}

// final / auxiliary heads: feat[l] = [lstm_in(9) | hs(5)]
__global__ void k_heads(const float* __restrict__ lstm_in, const float* __restrict__ hs,
                        const float* __restrict__ Wt, const float* __restrict__ bt,
                        const float* __restrict__ Wa, const float* __restrict__ ba,
                        float* __restrict__ out) {
  const int t = blockIdx.x * blockDim.x + threadIdx.x;
  if (t >= LAG * 2) return;
  const int l = t >> 1, cc = t & 1;
  float f = bt[cc], a = ba[cc];
#pragma unroll
  for (int x = 0; x < 9; ++x) {
    const float v = lstm_in[l * 9 + x];
    f += v * Wt[x * 2 + cc]; a += v * Wa[x * 2 + cc];
  }
#pragma unroll
  for (int j = 0; j < 5; ++j) {
    const float v = hs[l * 5 + j];
    f += v * Wt[(9 + j) * 2 + cc]; a += v * Wa[(9 + j) * 2 + cc];
  }
  out[l * 2 + cc] = f;                 // final
  out[LAG * 2 + l * 2 + cc] = a;       // auxiliary
}

// ------------------------------- launcher -----------------------------------

extern "C" void kernel_launch(void* const* d_in, const int* in_sizes, int n_in,
                              void* d_out, int out_size, void* d_ws, size_t ws_size,
                              hipStream_t stream) {
  (void)in_sizes; (void)n_in; (void)out_size; (void)ws_size;

  const float* text   = (const float*)d_in[0];
  const float* prices = (const float*)d_in[1];
  const float* pos    = (const float*)d_in[2];
  const float* Wsrc[10] = {
    (const float*)d_in[3],  // Wq1
    (const float*)d_in[4],  // Wk1
    (const float*)d_in[5],  // Wv1
    (const float*)d_in[6],  // Wo1
    (const float*)d_in[9],  // W1
    (const float*)d_in[11], // W2
    (const float*)d_in[13], // Wq2
    (const float*)d_in[14], // Wk2
    (const float*)d_in[15], // Wv2
    (const float*)d_in[16], // Wo2
  };
  const float* ln_g = (const float*)d_in[7];
  const float* ln_b = (const float*)d_in[8];
  const float* b1   = (const float*)d_in[10];
  const float* b2   = (const float*)d_in[12];
  const float* w_u  = (const float*)d_in[17];
  const float* w_m  = (const float*)d_in[18];
  const float* Wih  = (const float*)d_in[19];
  const float* Whh  = (const float*)d_in[20];
  const float* bih  = (const float*)d_in[21];
  const float* bhh  = (const float*)d_in[22];
  const float* Wt   = (const float*)d_in[23];
  const float* bt   = (const float*)d_in[24];
  const float* Wa   = (const float*)d_in[25];
  const float* ba   = (const float*)d_in[26];
  float* out = (float*)d_out;

  // workspace carving (needs ~341 MB)
  char* ws = (char*)d_ws;
  size_t off = 0;
  auto carve = [&](size_t bytes) -> void* {
    void* p = ws + off;
    off += (bytes + 255) & ~(size_t)255;
    return p;
  };

  bf16_t* Wbt[10];
  for (int i = 0; i < 10; ++i) Wbt[i] = (bf16_t*)carve((size_t)DIM * DIM * sizeof(bf16_t));
  const size_t actB = (size_t)MROWS * DIM * sizeof(bf16_t);   // 64 MiB
  bf16_t* B0 = (bf16_t*)carve(actB);
  bf16_t* B1 = (bf16_t*)carve(actB);
  bf16_t* B2 = (bf16_t*)carve(actB);
  bf16_t* B3 = (bf16_t*)carve(actB);
  bf16_t* B4 = (bf16_t*)carve(actB);
  float* attv    = (float*)carve(DIM * sizeof(float));
  float* lstm_in = (float*)carve((size_t)LAG * 9 * sizeof(float));
  float* hs      = (float*)carve((size_t)LAG * 5 * sizeof(float));

  // prepass: bf16 transposed weights + fused (text + pos_embed)
  for (int i = 0; i < 10; ++i)
    k_wt_bt<<<(DIM * DIM) / 256, 256, 0, stream>>>(Wsrc[i], Wbt[i]);
  k_add_pos<<<(MROWS * DIM) / 256, 256, 0, stream>>>(text, pos, B0);

  const dim3 gg(DIM / TN, MROWS / TM);    // 8 x 256 blocks
  // --- encoder block 1 ---
  k_gemm<<<gg, 256, 0, stream>>>(B0, Wbt[0], nullptr, B1, 0);   // Q1
  k_gemm<<<gg, 256, 0, stream>>>(B0, Wbt[1], nullptr, B2, 0);   // K1
  k_gemm<<<gg, 256, 0, stream>>>(B0, Wbt[2], nullptr, B3, 0);   // V1
  k_attn_mh<<<(LAG * HEADS) / 4, 128, 0, stream>>>(B1, B2, B3, B4);
  k_gemm<<<gg, 256, 0, stream>>>(B4, Wbt[3], nullptr, B1, 0);   // inter
  k_ln<<<MROWS / 8, 256, 0, stream>>>(B1, B0, ln_g, ln_b, B2);  // Y
  k_gemm<<<gg, 256, 0, stream>>>(B2, Wbt[4], b1, B3, 2);        // relu(Y@W1+b1)
  k_gemm<<<gg, 256, 0, stream>>>(B3, Wbt[5], b2, B0, 1);        // ffn
  // --- encoder block 2 (single head) ---
  k_gemm<<<gg, 256, 0, stream>>>(B0, Wbt[6], nullptr, B1, 0);   // Q2
  k_gemm<<<gg, 256, 0, stream>>>(B0, Wbt[7], nullptr, B2, 0);   // K2
  k_gemm<<<gg, 256, 0, stream>>>(B0, Wbt[8], nullptr, B4, 0);   // V2
  k_attn_sh<<<LAG, 256, 0, stream>>>(B1, B2, B4, B3);
  k_gemm<<<gg, 256, 0, stream>>>(B3, Wbt[9], nullptr, B1, 0);   // m
  // --- temporal head ---
  k_attvec<<<1, 1024, 0, stream>>>(w_u, w_m, attv);
  k_outk<<<MROWS / 8, 256, 0, stream>>>(B1, attv, lstm_in);
  k_prices<<<(LAG * 5 + 255) / 256, 256, 0, stream>>>(prices, lstm_in);
  k_lstm<<<1, 32, 0, stream>>>(lstm_in, Wih, Whh, bih, bhh, hs);
  k_heads<<<(LAG * 2 + 255) / 256, 256, 0, stream>>>(lstm_in, hs, Wt, bt, Wa, ba, out);
}
